// GCN_alignment_gate_att_47502338294135
// MI455X (gfx1250) — matrix operand order repaired and checked by reading, not compiled, not run on hardware
//
#include <hip/hip_runtime.h>
#include <hip/hip_bf16.h>

typedef __attribute__((ext_vector_type(16))) __bf16 v16bf;
typedef __attribute__((ext_vector_type(2)))  __bf16 v2bf;
typedef __attribute__((ext_vector_type(8)))  float  v8f;
typedef __attribute__((ext_vector_type(8)))  int    vi8;
typedef __attribute__((ext_vector_type(4)))  int    vi4;

#define DEV static __device__ __forceinline__

#if __has_builtin(__builtin_amdgcn_cvt_pk_bf16_f32)
DEV unsigned pack2(float f0, float f1) {
  v2bf p = __builtin_amdgcn_cvt_pk_bf16_f32(f0, f1);
  return __builtin_bit_cast(unsigned, p);
}
#else
// Native packed f32->bf16 convert (VOP3); clang builtin not declared on this
// toolchain, so emit the instruction directly.
DEV unsigned pack2(float f0, float f1) {
  unsigned r;
  asm("v_cvt_pk_bf16_f32 %0, %1, %2" : "=v"(r) : "v"(f0), "v"(f1));
  return r;
}
#endif

DEV __bf16 f2bf(float f) {
  unsigned r = pack2(f, 0.f);
  return __builtin_bit_cast(__bf16, (unsigned short)(r & 0xFFFFu));
}

// Async global -> LDS copy of 16 bytes per lane (CDNA5 VGLOBAL async path).
// VDST = LDS byte offset (low 32 bits of generic shared pointer), VADDR = 64b.
DEV void async_copy_b128(unsigned lds_off, const void* gaddr) {
  unsigned long long ga = (unsigned long long)gaddr;
  asm volatile("global_load_async_to_lds_b128 %0, %1, off"
               :: "v"(lds_off), "v"(ga) : "memory");
}
DEV void async_wait0() {
  asm volatile("s_wait_asynccnt 0x0" ::: "memory");
}

// A fragment 16x32 (MxK), source row-major bf16 [..][ldK]. Each lane's 16 bf16
// values are two runs of 4 consecutive dwords -> 2x b128 loads.
DEV v16bf frag_a_rm(const __bf16* base, int m0, int k0, int ldK, int lane) {
  const int m  = m0 + (lane & 15);
  const int off = (m * ldK + k0 + ((lane >> 4) << 3)) >> 1;  // dword index
  const int* pi = (const int*)base;
  vi4 lo = *(const vi4*)(pi + off);
  vi4 hi = *(const vi4*)(pi + off + 8);
  vi8 w;
  w[0] = lo[0]; w[1] = lo[1]; w[2] = lo[2]; w[3] = lo[3];
  w[4] = hi[0]; w[5] = hi[1]; w[6] = hi[2]; w[7] = hi[3];
  return __builtin_bit_cast(v16bf, w);
}

// B fragment 32x16 (KxN), source is B^T stored n-major: baseT[n][k] bf16 with
// row length ldK.  Two runs of 4 consecutive dwords per lane -> 2x b128 loads.
DEV v16bf frag_b_nm(const void* baseT, int k0, int n0, int ldK, int lane) {
  const int n  = n0 + (lane & 15);
  const int off = (n * ldK + k0 + ((lane >> 4) << 3)) >> 1;  // dword index
  const int* pi = (const int*)baseT;
  vi4 lo = *(const vi4*)(pi + off);
  vi4 hi = *(const vi4*)(pi + off + 8);
  vi8 w;
  w[0] = lo[0]; w[1] = lo[1]; w[2] = lo[2]; w[3] = lo[3];
  w[4] = hi[0]; w[5] = hi[1]; w[6] = hi[2]; w[7] = hi[3];
  return __builtin_bit_cast(v16bf, w);
}

// f32 C/D 16x16: VGPR v holds row (lane>=16 ? 8 : 0)+v, col = lane&15.
DEV void store_c(float* base, v8f c, int m0, int n0, int ld, int lane) {
  int m = m0 + (lane >> 4) * 8;
  int n = n0 + (lane & 15);
#pragma unroll
  for (int v = 0; v < 8; ++v) base[(m + v) * ld + n] = c[v];
}

#define WMMA_BF16(a, b, c) \
  __builtin_amdgcn_wmma_f32_16x16x32_bf16(false, (a), false, (b), (short)0, (c), false, false)

// ---------------------------------------------------------------------------
// Prep: Ep[n][kp] = pack(E[2kp][n], E[2kp+1][n])  -- embedding transposed,
// pair-packed bf16 (1 MB): GEMM B-fragments become direct b128 global loads.
// ---------------------------------------------------------------------------
__global__ __launch_bounds__(256) void k_prepE(const float* __restrict__ E,
                                               unsigned* __restrict__ Ep) {
  int i = blockIdx.x * 256 + threadIdx.x;      // 64 * 4096 total
  int n = i >> 12, kp = i & 4095;
  float e0 = E[(size_t)(2 * kp) * 64 + n];
  float e1 = E[(size_t)(2 * kp + 1) * 64 + n];
  Ep[(size_t)n * 4096 + kp] = pack2(e0, e1);
}

// Prep: conv weights pair-packed bf16, done once (not per fused block).
__global__ __launch_bounds__(256) void k_prepW(const float* __restrict__ c1,
                                               const float* __restrict__ c2,
                                               unsigned* __restrict__ W1p,
                                               unsigned* __restrict__ W2p) {
  int i = blockIdx.x * 256 + threadIdx.x;      // 8192 + 16384 total
  if (i < 8192) {
    W1p[i] = pack2(c1[2 * i], c1[2 * i + 1]);
  } else {
    int j = i - 8192;
    W2p[j] = pack2(c2[2 * j], c2[2 * j + 1]);
  }
}

// ---------------------------------------------------------------------------
// Kernel 1: layer1[8192,64] = A[8192,8192] @ E[8192,64]   (bf16 WMMA, f32 acc)
// Block: 256 threads (8 waves), 64 rows; wave = 16 rows x 32 cols.
// A staged in LDS (f32->bf16 pair-packed); B fragments straight from Ep (L2).
// ---------------------------------------------------------------------------
__global__ __launch_bounds__(256) void k_layer1(const float* __restrict__ A,
                                                const unsigned* __restrict__ Ep,
                                                float* __restrict__ L1) {
  __shared__ __bf16 sA[64][64];    // 8 KB, pair-packed row-major
  const int tid = threadIdx.x, lane = tid & 31, wave = tid >> 5;
  const int row0 = blockIdx.x * 64;
  const int ms = (wave >> 1) * 16;
  const int ns = (wave & 1) * 32;
  v8f acc0 = {0.f,0.f,0.f,0.f,0.f,0.f,0.f,0.f};
  v8f acc1 = acc0;

  const float4* A4 = reinterpret_cast<const float4*>(A);
  for (int kk = 0; kk < 8192; kk += 64) {
    if (kk + 64 < 8192) {
      const float* pf = &A[(size_t)(row0 + (tid >> 2)) * 8192 + kk + 64 + (tid & 3) * 16];
      __builtin_prefetch(pf, 0, 3);
    }
#pragma unroll
    for (int i = tid; i < 64 * 16; i += 256) {   // 64 rows x 16 float4
      int r = i >> 4, c = i & 15;
      float4 v = A4[(size_t)(row0 + r) * 2048 + (kk >> 2) + c];
      uint2 pv;
      pv.x = pack2(v.x, v.y);
      pv.y = pack2(v.z, v.w);
      reinterpret_cast<uint2*>(&sA[r][0])[c] = pv;   // ds_store_b64
    }
    __syncthreads();
#pragma unroll
    for (int ks = 0; ks < 64; ks += 32) {
      v16bf af = frag_a_rm(&sA[0][0], ms, ks, 64, lane);
      v16bf b0 = frag_b_nm(Ep, kk + ks, ns,      8192, lane);
      v16bf b1 = frag_b_nm(Ep, kk + ks, ns + 16, 8192, lane);
      acc0 = WMMA_BF16(af, b0, acc0);
      acc1 = WMMA_BF16(af, b1, acc1);
    }
    __syncthreads();
  }
  int m = row0 + ms + (lane >> 4) * 8;
  int n = lane & 15;
#pragma unroll
  for (int v = 0; v < 8; ++v) {
    L1[(size_t)(m + v) * 64 + ns + n]      = acc0[v];
    L1[(size_t)(m + v) * 64 + ns + 16 + n] = acc1[v];
  }
}

// ---------------------------------------------------------------------------
// Branch encoder: gating + t build + conv1/conv2 GEMMs (WMMA) + max-pool.
// ---------------------------------------------------------------------------
static __device__ void branch_encode(const float* Xe, const float* Ee,
                                     __bf16 (*tbuf)[128],
                                     const __bf16 (*w1b)[128],
                                     const __bf16 (*w2b)[256],
                                     float (*pm)[128], float* gbuf, float* outp,
                                     const float* gate_w, float gate_bias,
                                     const float* conv1_b, const float* conv2_b,
                                     int tid, int lane, int wave) {
  const float NEG = -3.4e38f;
  if (tid < 64) {
    float s = gate_bias;
    for (int h = 0; h < 64; ++h) s += Xe[tid * 64 + h] * gate_w[h];
    gbuf[tid] = 1.f / (1.f + __expf(-s));
  }
  __syncthreads();
  // t[l] = [ (Xe-Xs)^2 , Xe*Xs ],  Xs = (1-g)Xe + g*Ee  (pair-packed writes)
  for (int i = tid; i < 64 * 32; i += 256) {
    int l = i >> 5, hp = i & 31, h0 = hp * 2;
    float g   = gbuf[l];
    float xe0 = Xe[l * 64 + h0],     ee0 = Ee[l * 64 + h0];
    float xe1 = Xe[l * 64 + h0 + 1], ee1 = Ee[l * 64 + h0 + 1];
    float xs0 = (1.f - g) * xe0 + g * ee0, sub0 = xe0 - xs0;
    float xs1 = (1.f - g) * xe1 + g * ee1, sub1 = xe1 - xs1;
    unsigned* trow = (unsigned*)&tbuf[l][0];
    trow[hp]      = pack2(sub0 * sub0, sub1 * sub1);
    trow[32 + hp] = pack2(xe0 * xs0, xe1 * xs1);
  }
  if (tid < 64) ((unsigned*)&tbuf[64][0])[tid] = 0u;   // pad row for conv2
  __syncthreads();

  // e1 = t[64,128] @ conv1_w^T  -> tile-max into pm
  for (int t = wave * 4; t < wave * 4 + 4; ++t) {
    int mt = t >> 3, nt = t & 7;
    v8f acc = {0.f,0.f,0.f,0.f,0.f,0.f,0.f,0.f};
    for (int ks = 0; ks < 128; ks += 32) {
      v16bf af = frag_a_rm(&tbuf[0][0], mt * 16, ks, 128, lane);
      v16bf bf = frag_b_nm(&w1b[0][0], ks, nt * 16, 128, lane);
      acc = WMMA_BF16(af, bf, acc);
    }
    float mx = NEG;
#pragma unroll
    for (int v = 0; v < 8; ++v) mx = fmaxf(mx, acc[v]);
    mx = fmaxf(mx, __shfl_xor(mx, 16));
    if (lane < 16) pm[mt][nt * 16 + lane] = mx;
  }
  __syncthreads();
  if (tid < 128) {
    float mx = fmaxf(fmaxf(pm[0][tid], pm[1][tid]), fmaxf(pm[2][tid], pm[3][tid]));
    outp[tid] = fmaxf(mx + conv1_b[tid], 0.f);
  }
  __syncthreads();

  // e2 = [t[l],t[l+1]] (64x256, row 63 masked) @ conv2_w^T
  for (int t = wave * 4; t < wave * 4 + 4; ++t) {
    int mt = t >> 3, nt = t & 7;
    v8f acc = {0.f,0.f,0.f,0.f,0.f,0.f,0.f,0.f};
    for (int ks = 0; ks < 256; ks += 32) {
      int roff = ks >> 7, kb = ks & 127;    // second half reads row l+1
      v16bf af = frag_a_rm(&tbuf[roff][0], mt * 16, kb, 128, lane);
      v16bf bf = frag_b_nm(&w2b[0][0], ks, nt * 16, 256, lane);
      acc = WMMA_BF16(af, bf, acc);
    }
    float mx = NEG;
    int mbase = mt * 16 + (lane >> 4) * 8;
#pragma unroll
    for (int v = 0; v < 8; ++v)
      if (mbase + v != 63) mx = fmaxf(mx, acc[v]);
    mx = fmaxf(mx, __shfl_xor(mx, 16));
    if (lane < 16) pm[mt][nt * 16 + lane] = mx;
  }
  __syncthreads();
  if (tid < 128) {
    float mx = fmaxf(fmaxf(pm[0][tid], pm[1][tid]), fmaxf(pm[2][tid], pm[3][tid]));
    outp[128 + tid] = fmaxf(mx + conv2_b[tid], 0.f);
  }
  __syncthreads();
}

// ---------------------------------------------------------------------------
// Kernel 2: fused per-batch attention + gating + encode + dense + log_softmax.
// One workgroup (8 waves) per batch element; everything LDS-resident.
// ---------------------------------------------------------------------------
__global__ __launch_bounds__(256) void k_fused(
    const float* __restrict__ L1,
    const unsigned* __restrict__ W1p, const unsigned* __restrict__ W2p,
    const float* __restrict__ gate_w, const float* __restrict__ gate_b,
    const float* __restrict__ conv1_b, const float* __restrict__ conv2_b,
    const float* __restrict__ dense_w, const float* __restrict__ dense_b,
    const int* __restrict__ idx_q, const int* __restrict__ idx_a,
    float* __restrict__ out) {
  __shared__ float  Qf[64][64];
  __shared__ float  Af[64][64];
  __shared__ __bf16 Qb[64][64];     // row-major (A frags + B^T for score)
  __shared__ __bf16 Ab[64][64];
  __shared__ __bf16 QbT[64][64];    // [h][l] transposed (B for EA)
  __shared__ __bf16 AbT[64][64];    // [h][l] transposed (B for EQ)
  __shared__ float  S[64][64];      // score -> EQ -> EA
  __shared__ __bf16 wqb[64][64];
  __shared__ __bf16 wab[64][64];
  __shared__ __bf16 tbuf[65][128];
  __shared__ __bf16 w1b[128][128];  // [k_out][h]  (pair-packed == B^T)
  __shared__ __bf16 w2b[128][256];  // [k_out][j]
  __shared__ float  pm[4][128];
  __shared__ float  pq[256];
  __shared__ float  pa[256];
  __shared__ float  gbuf[64];
  __shared__ float  logits[2];

  const int b = blockIdx.x;
  const int tid = threadIdx.x, lane = tid & 31, wave = tid >> 5;

  // Async-stage prepacked conv weights into LDS (overlaps the gather below).
  {
    unsigned l1off = (unsigned)(size_t)&w1b[0][0];
    unsigned l2off = (unsigned)(size_t)&w2b[0][0];
    const char* g1 = (const char*)W1p;
    const char* g2 = (const char*)W2p;
    for (int i = tid; i < 2048; i += 256)     // 32 KB
      async_copy_b128(l1off + i * 16, g1 + (size_t)i * 16);
    for (int i = tid; i < 4096; i += 256)     // 64 KB
      async_copy_b128(l2off + i * 16, g2 + (size_t)i * 16);
  }
  // gather rows (+ transposed bf16 copies)
  for (int i = tid; i < 64 * 64; i += 256) {
    int l = i >> 6, h = i & 63;
    float q = L1[(size_t)idx_q[b * 64 + l] * 64 + h];
    float a = L1[(size_t)idx_a[b * 64 + l] * 64 + h];
    __bf16 qb = f2bf(q), ab = f2bf(a);
    Qf[l][h] = q; Qb[l][h] = qb; QbT[h][l] = qb;
    Af[l][h] = a; Ab[l][h] = ab; AbT[h][l] = ab;
  }
  async_wait0();
  __syncthreads();

  // score[q][a] = Q @ Aw^T   (B^T == Ab row-major)
  for (int t = wave * 2; t < wave * 2 + 2; ++t) {
    int mt = t >> 2, nt = t & 3;
    v8f acc = {0.f,0.f,0.f,0.f,0.f,0.f,0.f,0.f};
    for (int ks = 0; ks < 64; ks += 32) {
      v16bf af = frag_a_rm(&Qb[0][0], mt * 16, ks, 64, lane);
      v16bf bf = frag_b_nm(&Ab[0][0], ks, nt * 16, 64, lane);
      acc = WMMA_BF16(af, bf, acc);
    }
    store_c(&S[0][0], acc, mt * 16, nt * 16, 64, lane);
  }
  __syncthreads();

  if (tid < 64) {           // wq: row softmax over a (pair-packed writes)
    int q = tid; float mx = -3.4e38f;
    for (int c = 0; c < 64; ++c) mx = fmaxf(mx, S[q][c]);
    float sum = 0.f;
    for (int c = 0; c < 64; ++c) sum += __expf(S[q][c] - mx);
    float inv = 1.f / sum;
    unsigned* wrow = (unsigned*)&wqb[q][0];
    for (int c = 0; c < 64; c += 2)
      wrow[c >> 1] = pack2(__expf(S[q][c] - mx) * inv, __expf(S[q][c + 1] - mx) * inv);
  } else if (tid < 128) {   // wa: column softmax over q
    int a = tid - 64; float mx = -3.4e38f;
    for (int q = 0; q < 64; ++q) mx = fmaxf(mx, S[q][a]);
    float sum = 0.f;
    for (int q = 0; q < 64; ++q) sum += __expf(S[q][a] - mx);
    float inv = 1.f / sum;
    unsigned* wrow = (unsigned*)&wab[a][0];
    for (int q = 0; q < 64; q += 2)
      wrow[q >> 1] = pack2(__expf(S[q][a] - mx) * inv, __expf(S[q + 1][a] - mx) * inv);
  }
  __syncthreads();

  // EQ = wq @ Aw -> S   (B^T == AbT)
  for (int t = wave * 2; t < wave * 2 + 2; ++t) {
    int mt = t >> 2, nt = t & 3;
    v8f acc = {0.f,0.f,0.f,0.f,0.f,0.f,0.f,0.f};
    for (int ks = 0; ks < 64; ks += 32) {
      v16bf af = frag_a_rm(&wqb[0][0], mt * 16, ks, 64, lane);
      v16bf bf = frag_b_nm(&AbT[0][0], ks, nt * 16, 64, lane);
      acc = WMMA_BF16(af, bf, acc);
    }
    store_c(&S[0][0], acc, mt * 16, nt * 16, 64, lane);
  }
  __syncthreads();
  branch_encode(&Qf[0][0], &S[0][0], tbuf, w1b, w2b, pm, gbuf, pq,
                gate_w, gate_b[0], conv1_b, conv2_b, tid, lane, wave);

  // EA = wa @ Q -> S   (B^T == QbT)
  for (int t = wave * 2; t < wave * 2 + 2; ++t) {
    int mt = t >> 2, nt = t & 3;
    v8f acc = {0.f,0.f,0.f,0.f,0.f,0.f,0.f,0.f};
    for (int ks = 0; ks < 64; ks += 32) {
      v16bf af = frag_a_rm(&wab[0][0], mt * 16, ks, 64, lane);
      v16bf bf = frag_b_nm(&QbT[0][0], ks, nt * 16, 64, lane);
      acc = WMMA_BF16(af, bf, acc);
    }
    store_c(&S[0][0], acc, mt * 16, nt * 16, 64, lane);
  }
  __syncthreads();
  branch_encode(&Af[0][0], &S[0][0], tbuf, w1b, w2b, pm, gbuf, pa,
                gate_w, gate_b[0], conv1_b, conv2_b, tid, lane, wave);

  if (tid < 2) {
    float s = dense_b[tid];
    for (int i = 0; i < 256; ++i) s += pq[i] * dense_w[i * 2 + tid];
    for (int i = 0; i < 256; ++i) s += pa[i] * dense_w[(256 + i) * 2 + tid];
    logits[tid] = s;
  }
  __syncthreads();
  if (tid == 0) {
    float l0 = logits[0], l1 = logits[1];
    float m = fmaxf(l0, l1);
    float lse = m + __logf(__expf(l0 - m) + __expf(l1 - m));
    out[b * 2 + 0] = l0 - lse;
    out[b * 2 + 1] = l1 - lse;
  }
}

extern "C" void kernel_launch(void* const* d_in, const int* in_sizes, int n_in,
                              void* d_out, int out_size, void* d_ws, size_t ws_size,
                              hipStream_t stream) {
  (void)in_sizes; (void)n_in; (void)out_size; (void)ws_size;
  const float* A       = (const float*)d_in[0];
  const float* E       = (const float*)d_in[1];
  const float* gate_w  = (const float*)d_in[2];
  const float* gate_b  = (const float*)d_in[3];
  const float* conv1_w = (const float*)d_in[4];
  const float* conv1_b = (const float*)d_in[5];
  const float* conv2_w = (const float*)d_in[6];
  const float* conv2_b = (const float*)d_in[7];
  const float* dense_w = (const float*)d_in[8];
  const float* dense_b = (const float*)d_in[9];
  const int*   idx_q   = (const int*)d_in[10];
  const int*   idx_a   = (const int*)d_in[11];
  float* out = (float*)d_out;

  char* ws = (char*)d_ws;
  float*    L1  = (float*)ws;                                   // 2 MB
  unsigned* Ep  = (unsigned*)(ws + (size_t)8192 * 64 * 4);      // 1 MB
  unsigned* W1p = (unsigned*)(ws + 3 * 1024 * 1024);            // 32 KB
  unsigned* W2p = (unsigned*)(ws + 3 * 1024 * 1024 + 32 * 1024);// 64 KB

  k_prepE <<<(64 * 4096) / 256, 256, 0, stream>>>(E, Ep);
  k_prepW <<<(8192 + 16384) / 256, 256, 0, stream>>>(conv1_w, conv2_w, W1p, W2p);
  k_layer1<<<8192 / 64,          256, 0, stream>>>(A, Ep, L1);
  k_fused <<<512,                256, 0, stream>>>(L1, W1p, W2p, gate_w, gate_b,
                                                   conv1_b, conv2_b,
                                                   dense_w, dense_b,
                                                   idx_q, idx_a, out);
}
